// Attention_63247688401034
// MI455X (gfx1250) — matrix operand order repaired and checked by reading, not compiled, not run on hardware
//
#include <hip/hip_runtime.h>

// ---------------------------------------------------------------------------
// CDNA5 (gfx1250) wave32 WMMA attention: x@Wqkv+b -> flash-attn -> @Wproj+b
// All GEMMs in bf16 via v_wmma_f32_16x16x32_bf16 (f32 accumulate).
// Global->LDS tile staging via GLOBAL_LOAD_ASYNC_TO_LDS_B128 when available.
// ---------------------------------------------------------------------------

typedef __attribute__((ext_vector_type(16))) __bf16 v16bf;
typedef __attribute__((ext_vector_type(8)))  __bf16 v8bf;
typedef __attribute__((ext_vector_type(8)))  float  v8f;
typedef __attribute__((ext_vector_type(4)))  int    v4i;

#if __has_builtin(__builtin_amdgcn_global_load_async_to_lds_b128)
#define HAVE_ASYNC_LDS 1
typedef __attribute__((address_space(1))) v4i* as1_v4ip;
typedef __attribute__((address_space(3))) v4i* as3_v4ip;
#else
#define HAVE_ASYNC_LDS 0
#endif

__device__ __forceinline__ unsigned short f2bf(float f) {
  union { float f; unsigned u; } v; v.f = f;
  unsigned u = v.u + 0x7FFFu + ((v.u >> 16) & 1u);   // round-to-nearest-even
  return (unsigned short)(u >> 16);
}

// lo = 8 bf16 at p0, hi = 8 bf16 at p1 (both 16B aligned) -> one v16bf frag
__device__ __forceinline__ v16bf ld_frag(const unsigned short* p0,
                                         const unsigned short* p1) {
  v8bf lo = *(const v8bf*)p0;
  v8bf hi = *(const v8bf*)p1;
  return __builtin_shufflevector(lo, hi, 0,1,2,3,4,5,6,7,8,9,10,11,12,13,14,15);
}

__device__ __forceinline__ v8f wmma_bf16(v16bf a, v16bf b, v8f c) {
  return __builtin_amdgcn_wmma_f32_16x16x32_bf16(
      /*neg_a=*/false, a, /*neg_b=*/false, b,
      /*c_mod=*/(short)0, c, /*reuse_a=*/false, /*reuse_b=*/false);
}

// 16-byte global->LDS copy at compile-time byte offset OFF from both bases.
template <int OFF>
__device__ __forceinline__ void async_copy16(const unsigned short* g,
                                             unsigned short* l) {
#if HAVE_ASYNC_LDS
  unsigned short* gnc = const_cast<unsigned short*>(g);
  __builtin_amdgcn_global_load_async_to_lds_b128(
      (as1_v4ip)gnc, (as3_v4ip)l, OFF, 0);
#else
  *(v8bf*)((char*)l + OFF) = *(const v8bf*)((const char*)g + OFF);
#endif
}

__device__ __forceinline__ void wait_async() {
#if HAVE_ASYNC_LDS
#if __has_builtin(__builtin_amdgcn_s_wait_asynccnt)
  __builtin_amdgcn_s_wait_asynccnt(0);
#else
  asm volatile("s_wait_asynccnt 0x0" ::: "memory");
#endif
#endif
}

// ---------------------------------------------------------------------------
// Elementwise fp32 -> bf16
// ---------------------------------------------------------------------------
__global__ __launch_bounds__(256) void cvt_bf16(const float* __restrict__ in,
                                                unsigned short* __restrict__ out,
                                                size_t n) {
  size_t i = ((size_t)blockIdx.x * 256 + threadIdx.x) * 4;
  if (i + 3 >= n) return;
  float4 f = *(const float4*)(in + i);
  out[i + 0] = f2bf(f.x);
  out[i + 1] = f2bf(f.y);
  out[i + 2] = f2bf(f.z);
  out[i + 3] = f2bf(f.w);
}

// in: [K][N] fp32  ->  out: [N][K] bf16   (gridDim = (K, N/256))
__global__ __launch_bounds__(256) void transpose_bf16(const float* __restrict__ in,
                                                      unsigned short* __restrict__ out,
                                                      int K, int N) {
  int k = blockIdx.x;
  int n = blockIdx.y * 256 + threadIdx.x;
  out[(size_t)n * K + k] = f2bf(in[(size_t)k * N + n]);
}

// ---------------------------------------------------------------------------
// GEMM: C[M,N] = A[M,K](bf16) @ Bt[N,K](bf16)^T + bias[N](f32)
// BM=128 BN=128 BK=64, 256 threads = 8 waves in 4x2; wave tile 32x64.
// ---------------------------------------------------------------------------
#define GLDT 72   // padded LDS stride (elements): 144 B, multiple of 16 B

template <bool OUT_F32>
__global__ __launch_bounds__(256) void gemm_wmma(
    const unsigned short* __restrict__ A,
    const unsigned short* __restrict__ Bt,
    const float* __restrict__ bias,
    void* __restrict__ Cv,
    int M, int N, int K) {
  __shared__ unsigned short sA[128 * GLDT];
  __shared__ unsigned short sB[128 * GLDT];

  const int tid  = threadIdx.x;
  const int lane = tid & 31;
  const int wave = tid >> 5;
  const int wm   = wave & 3;          // 0..3 (row blocks of 32)
  const int wn   = wave >> 2;         // 0..1 (col blocks of 64)
  const int half = lane >> 4;
  const int lrow16 = lane & 15;
  const int m0 = blockIdx.y * 128;
  const int n0 = blockIdx.x * 128;

  v8f acc[2][4];
#pragma unroll
  for (int mi = 0; mi < 2; ++mi)
#pragma unroll
    for (int ni = 0; ni < 4; ++ni)
      acc[mi][ni] = (v8f){0,0,0,0,0,0,0,0};

  const int trow = tid >> 1;          // 0..127
  const int tcol = (tid & 1) * 32;    // 0 / 32

  for (int k0 = 0; k0 < K; k0 += 64) {
    __syncthreads();
    {
      const unsigned short* ga = A  + (size_t)(m0 + trow) * K + k0 + tcol;
      const unsigned short* gb = Bt + (size_t)(n0 + trow) * K + k0 + tcol;
      unsigned short* la = sA + trow * GLDT + tcol;
      unsigned short* lb = sB + trow * GLDT + tcol;
      async_copy16<0>(ga, la);  async_copy16<16>(ga, la);
      async_copy16<32>(ga, la); async_copy16<48>(ga, la);
      async_copy16<0>(gb, lb);  async_copy16<16>(gb, lb);
      async_copy16<32>(gb, lb); async_copy16<48>(gb, lb);
      if (k0 + 64 < K) {
        __builtin_prefetch(ga + 64, 0, 0);
        __builtin_prefetch(gb + 64, 0, 0);
      }
    }
    wait_async();
    __syncthreads();

#pragma unroll
    for (int s = 0; s < 2; ++s) {      // two K=32 WMMA steps per tile
      v16bf af[2], bfr[4];
      const int ka = s * 32 + half * 8;    // A-layout: halves own K 0-7 / 8-15
      const int kb = s * 32 + half * 8;    // B-layout: same as A (halves own K 0-7,16-23 / 8-15,24-31)
#pragma unroll
      for (int mi = 0; mi < 2; ++mi) {
        const unsigned short* p = sA + (wm * 32 + mi * 16 + lrow16) * GLDT + ka;
        af[mi] = ld_frag(p, p + 16);
      }
#pragma unroll
      for (int ni = 0; ni < 4; ++ni) {
        const unsigned short* p = sB + (wn * 64 + ni * 16 + lrow16) * GLDT + kb;
        bfr[ni] = ld_frag(p, p + 16);
      }
#pragma unroll
      for (int mi = 0; mi < 2; ++mi)
#pragma unroll
        for (int ni = 0; ni < 4; ++ni)
          acc[mi][ni] = wmma_bf16(af[mi], bfr[ni], acc[mi][ni]);
    }
  }

  // epilogue: C element (m = base + r + 8*half, n = lane&15 base)
#pragma unroll
  for (int mi = 0; mi < 2; ++mi) {
    const int rbase = m0 + wm * 32 + mi * 16 + half * 8;
#pragma unroll
    for (int ni = 0; ni < 4; ++ni) {
      const int col = n0 + wn * 64 + ni * 16 + lrow16;
      const float bv = bias[col];
#pragma unroll
      for (int r = 0; r < 8; ++r) {
        const float v = acc[mi][ni][r] + bv;
        if (OUT_F32)
          ((float*)Cv)[(size_t)(rbase + r) * N + col] = v;
        else
          ((unsigned short*)Cv)[(size_t)(rbase + r) * N + col] = f2bf(v);
      }
    }
  }
}

// ---------------------------------------------------------------------------
// Flash attention: qkv[B*S, 3072] bf16 -> out[B*S, 1024] bf16 (heads merged)
// 128 threads = 4 waves; block = 64 q rows of one (b,h); wave = 16 q rows.
// ---------------------------------------------------------------------------
#define ALDT 72

__global__ __launch_bounds__(128) void flash_attn(
    const unsigned short* __restrict__ qkv,
    unsigned short* __restrict__ out,
    int Bb, int S, int H) {
  __shared__ unsigned short sK [64 * ALDT];   // [key][d]
  __shared__ unsigned short sVT[64 * ALDT];   // [d][key]
  __shared__ unsigned short sP [64 * ALDT];   // 4 waves x 16 rows

  const int tid  = threadIdx.x;
  const int lane = tid & 31;
  const int wave = tid >> 5;
  const int half = lane >> 4;
  const int ncol = lane & 15;

  const int qt = blockIdx.x & 15;         // S/64 = 16 q-tiles
  const int bh = blockIdx.x >> 4;
  const int h  = bh & 15;                 // H = 16
  const int b  = bh >> 4;

  const int D3   = 3 * H * 64;            // 3072
  const int row0 = b * S;

  // Q fragments: A-layout straight from global (row-contiguous d)
  const int qrow = qt * 64 + wave * 16 + ncol;   // this lane's q row (A side)
  v16bf qf[2];
  {
    const unsigned short* qp = qkv + (size_t)(row0 + qrow) * D3 + h * 64;
#pragma unroll
    for (int s = 0; s < 2; ++s) {
      const int d0 = s * 32 + half * 8;
      qf[s] = ld_frag(qp + d0, qp + d0 + 16);
    }
  }

  v8f o[4];
#pragma unroll
  for (int ni = 0; ni < 4; ++ni) o[ni] = (v8f){0,0,0,0,0,0,0,0};
  float mrow[8], lsum[8];
#pragma unroll
  for (int r = 0; r < 8; ++r) { mrow[r] = -1e30f; lsum[r] = 0.0f; }

  const int trow = tid >> 1;              // 0..63
  const int tcol = (tid & 1) * 32;        // 0 / 32
  unsigned short* pw = sP + wave * 16 * ALDT;

  for (int j = 0; j <= qt; ++j) {
    __syncthreads();
    {
      const size_t grow = (size_t)(row0 + j * 64 + trow) * D3 + h * 64 + tcol;
      const unsigned short* kp = qkv + grow + 1024;        // K section
      const unsigned short* vp = qkv + grow + 2048;        // V section
      unsigned short* lk = sK + trow * ALDT + tcol;
      async_copy16<0>(kp, lk);  async_copy16<16>(kp, lk);  // K tile: async
      async_copy16<32>(kp, lk); async_copy16<48>(kp, lk);
      union { v8bf v[4]; unsigned short u[32]; } tv;        // V tile: transpose
#pragma unroll
      for (int i = 0; i < 4; ++i) tv.v[i] = *(const v8bf*)(vp + 8 * i);
#pragma unroll
      for (int i = 0; i < 32; ++i)
        sVT[(tcol + i) * ALDT + trow] = tv.u[i];
    }
    wait_async();
    __syncthreads();

    // ---- S = Q @ K^T -------------------------------------------------------
    v8f sc[4];
#pragma unroll
    for (int ni = 0; ni < 4; ++ni) sc[ni] = (v8f){0,0,0,0,0,0,0,0};
#pragma unroll
    for (int st = 0; st < 2; ++st) {
      const int db = st * 32 + half * 8;
#pragma unroll
      for (int ni = 0; ni < 4; ++ni) {
        const unsigned short* p = sK + (ni * 16 + ncol) * ALDT + db;
        sc[ni] = wmma_bf16(qf[st], ld_frag(p, p + 16), sc[ni]);
      }
    }

    // ---- scale, causal mask, online softmax (rows private to wave-half) ---
    float corr[8];
#pragma unroll
    for (int r = 0; r < 8; ++r) {
      const int qg = qt * 64 + wave * 16 + half * 8 + r;   // C-layout row
      float rmax = -1e30f;
#pragma unroll
      for (int ni = 0; ni < 4; ++ni) {
        const int key = j * 64 + ni * 16 + ncol;
        float v = sc[ni][r] * 0.125f;                      // 1/sqrt(64)
        if (key > qg) v = -10000.0f;                       // match reference
        sc[ni][r] = v;
        rmax = fmaxf(rmax, v);
      }
#pragma unroll
      for (int m = 8; m >= 1; m >>= 1)                     // 16-lane butterfly
        rmax = fmaxf(rmax, __shfl_xor(rmax, m, 32));
      const float mnew = fmaxf(mrow[r], rmax);
      const float c = __expf(mrow[r] - mnew);
      mrow[r] = mnew;
      corr[r] = c;
      float rs = 0.0f;
#pragma unroll
      for (int ni = 0; ni < 4; ++ni) {
        const float p = __expf(sc[ni][r] - mnew);
        sc[ni][r] = p;
        rs += p;
      }
#pragma unroll
      for (int m = 8; m >= 1; m >>= 1) rs += __shfl_xor(rs, m, 32);
      lsum[r] = lsum[r] * c + rs;
    }
#pragma unroll
    for (int ni = 0; ni < 4; ++ni)
#pragma unroll
      for (int r = 0; r < 8; ++r) o[ni][r] *= corr[r];

    // P: C-layout -> LDS -> A-layout
#pragma unroll
    for (int ni = 0; ni < 4; ++ni)
#pragma unroll
      for (int r = 0; r < 8; ++r)
        pw[(half * 8 + r) * ALDT + ni * 16 + ncol] = f2bf(sc[ni][r]);
    __syncthreads();

    // ---- O += P @ V --------------------------------------------------------
#pragma unroll
    for (int st = 0; st < 2; ++st) {
      const int ka = st * 32 + half * 8;
      const unsigned short* pp = pw + ncol * ALDT + ka;
      const v16bf pf = ld_frag(pp, pp + 16);
      const int kv = st * 32 + half * 8;
#pragma unroll
      for (int ni = 0; ni < 4; ++ni) {
        const unsigned short* vp = sVT + (ni * 16 + ncol) * ALDT + kv;
        o[ni] = wmma_bf16(pf, ld_frag(vp, vp + 16), o[ni]);
      }
    }
  }

  // ---- normalize + store (merged heads, bf16) ------------------------------
  float inv[8];
#pragma unroll
  for (int r = 0; r < 8; ++r) inv[r] = 1.0f / lsum[r];
#pragma unroll
  for (int ni = 0; ni < 4; ++ni)
#pragma unroll
    for (int r = 0; r < 8; ++r) {
      const int srow = qt * 64 + wave * 16 + half * 8 + r;
      out[(size_t)(row0 + srow) * (H * 64) + h * 64 + ni * 16 + ncol] =
          f2bf(o[ni][r] * inv[r]);
    }
}

// ---------------------------------------------------------------------------
// Launch
// ---------------------------------------------------------------------------
extern "C" void kernel_launch(void* const* d_in, const int* in_sizes, int n_in,
                              void* d_out, int out_size, void* d_ws, size_t ws_size,
                              hipStream_t stream) {
  (void)in_sizes; (void)n_in; (void)out_size; (void)ws_size;
  const float* x      = (const float*)d_in[0];
  const float* w_attn = (const float*)d_in[1];
  const float* b_attn = (const float*)d_in[2];
  const float* w_proj = (const float*)d_in[3];
  const float* b_proj = (const float*)d_in[4];
  float* out = (float*)d_out;

  const int Bb = 4, S = 1024, D = 1024, H = 16;
  const int M = Bb * S;                       // 4096

  char* ws = (char*)d_ws;
  unsigned short* xb   = (unsigned short*)(ws);                              //  8 MB
  unsigned short* waT  = (unsigned short*)(ws + (size_t) 8 * 1024 * 1024);   //  6 MB
  unsigned short* wpT  = (unsigned short*)(ws + (size_t)14 * 1024 * 1024);   //  2 MB
  unsigned short* qkv  = (unsigned short*)(ws + (size_t)16 * 1024 * 1024);   // 24 MB
  unsigned short* attn = (unsigned short*)(ws + (size_t)40 * 1024 * 1024);   //  8 MB

  cvt_bf16<<<(int)((size_t)M * D / 1024), 256, 0, stream>>>(x, xb, (size_t)M * D);
  transpose_bf16<<<dim3(D, 3 * D / 256), 256, 0, stream>>>(w_attn, waT, D, 3 * D);
  transpose_bf16<<<dim3(D, D / 256), 256, 0, stream>>>(w_proj, wpT, D, D);

  gemm_wmma<false><<<dim3(3 * D / 128, M / 128), 256, 0, stream>>>(
      xb, waT, b_attn, (void*)qkv, M, 3 * D, D);

  flash_attn<<<dim3(Bb * H * (S / 64)), 128, 0, stream>>>(qkv, attn, Bb, S, H);

  gemm_wmma<true><<<dim3(D / 128, M / 128), 256, 0, stream>>>(
      attn, wpT, b_proj, (void*)out, M, D, D);
}
